// ConvLayer_49795850830364
// MI455X (gfx1250) — matrix operand order repaired and checked by reading, not compiled, not run on hardware
//
#include <hip/hip_runtime.h>
#include <math.h>

// ---------------------------------------------------------------------------
// Types for CDNA5 WMMA (gfx1250, wave32)
// ---------------------------------------------------------------------------
typedef __attribute__((ext_vector_type(16))) __bf16 v16bf;
typedef __attribute__((ext_vector_type(8)))  __bf16 v8bf;
typedef __attribute__((ext_vector_type(8)))  float  v8f;
typedef int vint4 __attribute__((vector_size(16)));   // GCC-vector int4 (builtin param type)

#define NN 8192   // nodes
#define DD 128    // feature dim
#define KK 128    // clusters

// ---- async global->LDS copy (CDNA5), guarded with fallback ----------------
#if defined(__has_builtin)
#if __has_builtin(__builtin_amdgcn_global_load_async_to_lds_b128)
#define HAVE_ASYNC_LDS 1
#endif
#if __has_builtin(__builtin_amdgcn_s_wait_asynccnt)
#define HAVE_WAIT_ASYNC 1
#endif
#endif

typedef __attribute__((address_space(1))) vint4* as1_int4p;
typedef __attribute__((address_space(3))) vint4* as3_int4p;

__device__ inline void async_copy16(const void* g, void* l) {
#ifdef HAVE_ASYNC_LDS
    __builtin_amdgcn_global_load_async_to_lds_b128((as1_int4p)g, (as3_int4p)l, 0, 0);
#else
    *(vint4*)l = *(const vint4*)g;
#endif
}

__device__ inline void wait_async_copies() {
#ifdef HAVE_ASYNC_LDS
#ifdef HAVE_WAIT_ASYNC
    __builtin_amdgcn_s_wait_asynccnt(0);
#else
    asm volatile("s_wait_asynccnt 0x0" ::: "memory");
#endif
#endif
}

__device__ inline v8f wmma_bf16(v16bf a, v16bf b, v8f c) {
    // D(f32 16x16) = A(16x32 bf16) * B(32x16 bf16) + C
    return __builtin_amdgcn_wmma_f32_16x16x32_bf16(
        /*neg_a=*/false, a, /*neg_b=*/false, b,
        /*c_mod=*/(short)0, c, /*reuse_a=*/false, /*reuse_b=*/false);
}

// A-matrix (16x32 bf16, row-major source, ld elements):
// lane l (0..15) covers row M=l; hi = lane>>4 selects K groups.
// element i<8  -> K = k0 + hi*8 + i        (8 contiguous)
// element i>=8 -> K = k0 + 16 + hi*8 + i-8 (8 contiguous)
__device__ inline v16bf load_a_frag(const __bf16* base, int ld, int row, int k0,
                                    int l, int hi) {
    const __bf16* p0 = base + (row + l) * ld + k0 + hi * 8;
    v8bf lo = *(const v8bf*)p0;
    v8bf hh = *(const v8bf*)(p0 + 16);
    v16bf r;
#pragma unroll
    for (int i = 0; i < 8; ++i) { r[i] = lo[i]; r[8 + i] = hh[i]; }
    return r;
}

// A-fragment read with transpose: logical A[m,k] = S[k,m] (S row-major, ld).
__device__ inline v16bf load_a_frag_T(const __bf16* S, int ld, int k0, int m0,
                                      int l, int hi) {
    v16bf r;
#pragma unroll
    for (int i = 0; i < 16; ++i) {
        int kk = (i < 8) ? (hi * 8 + i) : (16 + hi * 8 + (i - 8));
        r[i] = S[(k0 + kk) * ld + m0 + l];
    }
    return r;
}

// B-matrix (32x16 bf16, row-major source): lane covers column N=n0+(lane&15),
// elements i follow the same K order as the A fragment (hi = lane>>4).
__device__ inline v16bf load_b_frag(const __bf16* base, int ld, int k0, int n0,
                                    int lane) {
    const int hi = lane >> 4;
    const __bf16* p = base + k0 * ld + n0 + (lane & 15);
    v16bf r;
#pragma unroll
    for (int i = 0; i < 16; ++i) {
        int kk = (i < 8) ? (hi * 8 + i) : (16 + hi * 8 + (i - 8));
        r[i] = p[kk * ld];
    }
    return r;
}

__device__ inline float mishf(float x) {
    float sp = (x > 20.f) ? x : log1pf(expf(x));
    return x * tanhf(sp);
}

// ---------------------------------------------------------------------------
// Edge kernels
// ---------------------------------------------------------------------------
__global__ void count_kernel(const int* __restrict__ src, const int* __restrict__ dst,
                             float* cnt, float* deg, int E) {
    int e = blockIdx.x * 256 + threadIdx.x;
    if (e < E) {
        atomicAdd(&cnt[dst[e]], 1.0f);
        atomicAdd(&deg[src[e]], 1.0f);
    }
}

// One wave per edge: lane handles 4 consecutive floats of the 128-dim feature.
__global__ void scatter_kernel(const float* __restrict__ x,
                               const int* __restrict__ src,
                               const int* __restrict__ dst,
                               float* agg, int E) {
    int t = blockIdx.x * 256 + threadIdx.x;
    int e = t >> 5;
    int lane = t & 31;
    if (e >= E) return;
    int s = src[e], d = dst[e];
    float4 v = ((const float4*)(x + (size_t)s * DD))[lane];
    float* o = agg + (size_t)d * DD + lane * 4;
    atomicAdd(o + 0, v.x);
    atomicAdd(o + 1, v.y);
    atomicAdd(o + 2, v.z);
    atomicAdd(o + 3, v.w);
}

// tr += sum_edges dot(s[src], s[dst])  (== trace(s^T A s) for multigraph adj)
__global__ __launch_bounds__(256) void trace_kernel(const float* __restrict__ S,
                                                    const int* __restrict__ src,
                                                    const int* __restrict__ dst,
                                                    float* tr, int E) {
    __shared__ float red[256];
    int wave = threadIdx.x >> 5, lane = threadIdx.x & 31;
    int e = blockIdx.x * 8 + wave;
    float p = 0.f;
    if (e < E) {
        const float4* a = (const float4*)(S + (size_t)src[e] * KK);
        const float4* b = (const float4*)(S + (size_t)dst[e] * KK);
        float4 xa = a[lane], xb = b[lane];
        p = xa.x * xb.x + xa.y * xb.y + xa.z * xb.z + xa.w * xb.w;
    }
    red[threadIdx.x] = p;
    __syncthreads();
    for (int s2 = 128; s2 > 0; s2 >>= 1) {
        if (threadIdx.x < s2) red[threadIdx.x] += red[threadIdx.x + s2];
        __syncthreads();
    }
    if (threadIdx.x == 0) atomicAdd(tr, red[0]);
}

// ---------------------------------------------------------------------------
// Fused dual GEMM: Y = ACT( (A1 [/cnt]) @ W1 + [A2 @ W2] + bias )
//   A1,A2: [8192x128] f32; W1,W2: [128x128] bf16 row-major; bias f32[128]
// 64 rows/block, 8 waves; wave w owns column tile w; 4 M-tiles per wave.
// Weights DMA'd global->LDS asynchronously while VALU converts A tiles.
// ---------------------------------------------------------------------------
template <bool DUAL, bool NORM, int ACT>
__global__ __launch_bounds__(256) void gemm_dual_kernel(
    const float* __restrict__ A1, const float* __restrict__ cnt,
    const float* __restrict__ A2,
    const __bf16* __restrict__ W1, const __bf16* __restrict__ W2,
    const float* __restrict__ bias, float* __restrict__ Y) {
    __shared__ __align__(16) __bf16 lA1[64 * DD];
    __shared__ __align__(16) __bf16 lA2[DUAL ? 64 * DD : 8];
    __shared__ __align__(16) __bf16 lW1[DD * DD];
    __shared__ __align__(16) __bf16 lW2[DUAL ? DD * DD : 8];
    const int tid = threadIdx.x;
    const int row0 = blockIdx.x * 64;

    // 1) kick off async DMA of weights into LDS (16B per lane per op)
#pragma unroll
    for (int j = 0; j < 8; ++j) {
        int c = (tid + 256 * j) * 8;              // bf16 element index
        async_copy16(W1 + c, lW1 + c);
        if (DUAL) async_copy16(W2 + c, lW2 + c);
    }
    // 2) overlap: stage A tiles with f32->bf16 conversion (VALU + ds stores)
    for (int i = tid; i < 64 * DD; i += 256) {
        int r = i >> 7;
        float sc = NORM ? (1.0f / fmaxf(cnt[row0 + r], 1.0f)) : 1.0f;
        lA1[i] = (__bf16)(A1[(size_t)row0 * DD + i] * sc);
        if (DUAL) lA2[i] = (__bf16)A2[(size_t)row0 * DD + i];
    }
    wait_async_copies();
    __syncthreads();

    const int w = tid >> 5, lane = tid & 31;
    const int l = lane & 15, hi = lane >> 4;
    const int n0 = w * 16;

    v16bf b1[4], b2[4];
#pragma unroll
    for (int kc = 0; kc < 4; ++kc) b1[kc] = load_b_frag(lW1, DD, kc * 32, n0, lane);
    if (DUAL) {
#pragma unroll
        for (int kc = 0; kc < 4; ++kc) b2[kc] = load_b_frag(lW2, DD, kc * 32, n0, lane);
    }
    const float bv = bias[n0 + l];

#pragma unroll
    for (int mt = 0; mt < 4; ++mt) {
        // batch-preload all A fragments for this tile (loads overlap)
        v16bf a1f[4];
#pragma unroll
        for (int kc = 0; kc < 4; ++kc)
            a1f[kc] = load_a_frag(lA1, DD, mt * 16, kc * 32, l, hi);
        v8f acc = {bv, bv, bv, bv, bv, bv, bv, bv};
#pragma unroll
        for (int kc = 0; kc < 4; ++kc) acc = wmma_bf16(a1f[kc], b1[kc], acc);
        if (DUAL) {
            v16bf a2f[4];
#pragma unroll
            for (int kc = 0; kc < 4; ++kc)
                a2f[kc] = load_a_frag(lA2, DD, mt * 16, kc * 32, l, hi);
            v8f acc2 = {};
#pragma unroll
            for (int kc = 0; kc < 4; ++kc) acc2 = wmma_bf16(a2f[kc], b2[kc], acc2);
            acc = acc + acc2;   // two independent WMMA chains merged here
        }
#pragma unroll
        for (int r = 0; r < 8; ++r) {
            float v = acc[r];
            if (ACT == 1) v = mishf(v);
            Y[(size_t)(row0 + mt * 16 + r + 8 * hi) * DD + n0 + l] = v;
        }
    }
}

// ---------------------------------------------------------------------------
// Split-K WMMA: out[128x128] += S_chunk^T @ B_chunk   (chunk = 64 rows/block)
// ---------------------------------------------------------------------------
__global__ __launch_bounds__(256) void pool_kernel(const float* __restrict__ S,
                                                   const float* __restrict__ B,
                                                   float* out) {
    __shared__ __align__(16) __bf16 lS[64 * KK];
    __shared__ __align__(16) __bf16 lB[64 * DD];
    int row0 = blockIdx.x * 64;
    for (int i = threadIdx.x; i < 64 * DD; i += 256) {
        lS[i] = (__bf16)S[(size_t)row0 * KK + i];
        lB[i] = (__bf16)B[(size_t)row0 * DD + i];
    }
    __syncthreads();

    int w = threadIdx.x >> 5, lane = threadIdx.x & 31;
    int l = lane & 15, hi = lane >> 4;
    int mt = w;  // cluster tile (8 tiles of 16 across K=128)
    // transposed A fragments are reused across all 8 nt tiles
    v16bf a0 = load_a_frag_T(lS, KK, 0, mt * 16, l, hi);
    v16bf a1 = load_a_frag_T(lS, KK, 32, mt * 16, l, hi);
    for (int nt = 0; nt < 8; ++nt) {
        v16bf bb0 = load_b_frag(lB, DD, 0, nt * 16, lane);
        v16bf bb1 = load_b_frag(lB, DD, 32, nt * 16, lane);
        v8f acc = {};
        v8f acc2 = {};
        acc = wmma_bf16(a0, bb0, acc);    // independent chains
        acc2 = wmma_bf16(a1, bb1, acc2);
        acc = acc + acc2;
#pragma unroll
        for (int r = 0; r < 8; ++r)
            atomicAdd(&out[(size_t)(mt * 16 + r + 8 * hi) * DD + nt * 16 + l], acc[r]);
    }
}

// ---------------------------------------------------------------------------
// Small kernels
// ---------------------------------------------------------------------------
__global__ void convert_bf16_kernel(const float* __restrict__ in, __bf16* out, int n) {
    int i = blockIdx.x * 256 + threadIdx.x;
    if (i < n) out[i] = (__bf16)in[i];
}

__global__ void add_kernel(const float* __restrict__ A, float* B, int n) {
    int i = blockIdx.x * 256 + threadIdx.x;
    if (i < n) B[i] = A[i] + B[i];
}

__global__ void softmax_kernel(float* X) {
    __shared__ float red[KK];
    int r = blockIdx.x, t = threadIdx.x;
    float v = X[(size_t)r * KK + t];
    red[t] = v;
    __syncthreads();
    for (int s2 = 64; s2 > 0; s2 >>= 1) {
        if (t < s2) red[t] = fmaxf(red[t], red[t + s2]);
        __syncthreads();
    }
    float mx = red[0];
    __syncthreads();
    float ex = expf(v - mx);
    red[t] = ex;
    __syncthreads();
    for (int s2 = 64; s2 > 0; s2 >>= 1) {
        if (t < s2) red[t] += red[t + s2];
        __syncthreads();
    }
    X[(size_t)r * KK + t] = ex / red[0];
}

// colsum[k] += sum_n s[n,k];  ca[k] += sum_n s[n,k]*deg[n]   (chunked atomics)
__global__ void colsum_ca_kernel(const float* __restrict__ S,
                                 const float* __restrict__ deg,
                                 float* colsum, float* ca) {
    int k = threadIdx.x;
    int row0 = blockIdx.x * 128;
    float cs = 0.f, cacc = 0.f;
    for (int r = 0; r < 128; ++r) {
        float v = S[(size_t)(row0 + r) * KK + k];
        cs += v;
        cacc += v * deg[row0 + r];
    }
    atomicAdd(&colsum[k], cs);
    atomicAdd(&ca[k], cacc);
}

__device__ inline float block_sum256(float v, float* red) {
    int t = threadIdx.x;
    red[t] = v;
    __syncthreads();
    for (int s2 = 128; s2 > 0; s2 >>= 1) {
        if (t < s2) red[t] += red[t + s2];
        __syncthreads();
    }
    float r = red[0];
    __syncthreads();
    return r;
}

// Final epilogue: SELU(pooled) -> out, and spectral+ortho+cluster losses.
__global__ __launch_bounds__(256) void final_kernel(
    const float* pooled1, const float* ss1, const float* ca1,
    const float* colsum1, const float* tr1,
    const float* pooled2, const float* ss2, const float* ca2,
    const float* colsum2, const float* tr2,
    float* out, int E) {
    __shared__ float red[256];
    int t = threadIdx.x;
    float loss = 0.f;
    const float twoM = (float)E;            // 2*m = total degree = E
    const float sqrtK = sqrtf((float)KK);   // ||I_K||_F
    for (int g = 0; g < 2; ++g) {
        const float* pooled = g ? pooled2 : pooled1;
        const float* ss     = g ? ss2     : ss1;
        const float* ca     = g ? ca2     : ca1;
        const float* colsum = g ? colsum2 : colsum1;
        const float* tr     = g ? tr2     : tr1;
        for (int i = t; i < KK * DD; i += 256) {
            float v = pooled[i];
            float s = (v > 0.f) ? v : 1.6732632423543772f * (expf(v) - 1.f);
            out[g * KK * DD + i] = 1.0507009873554805f * s;
        }
        float ca2sum = block_sum256((t < KK) ? ca[t] * ca[t] : 0.f, red);
        float spectral = -(tr[0] - ca2sum / twoM) / twoM;
        float sq = 0.f;
        for (int i = t; i < KK * KK; i += 256) { float v = ss[i]; sq += v * v; }
        float nF = sqrtf(block_sum256(sq, red));
        float dsum = 0.f;
        for (int i = t; i < KK * KK; i += 256) {
            float v = ss[i] / nF - (((i >> 7) == (i & 127)) ? (1.f / sqrtK) : 0.f);
            dsum += v * v;
        }
        float ortho = sqrtf(block_sum256(dsum, red));
        float cs2 = block_sum256((t < KK) ? colsum[t] * colsum[t] : 0.f, red);
        float cluster = sqrtf(cs2) / (float)NN * sqrtK - 1.f;
        loss += spectral + ortho + cluster;
    }
    if (t == 0) out[2 * KK * DD] = loss;
}

// ---------------------------------------------------------------------------
// Host orchestration
// ---------------------------------------------------------------------------
extern "C" void kernel_launch(void* const* d_in, const int* in_sizes, int n_in,
                              void* d_out, int out_size, void* d_ws, size_t ws_size,
                              hipStream_t stream) {
    const float* x1 = (const float*)d_in[0];
    const float* x2 = (const float*)d_in[1];
    const int*   e1 = (const int*)d_in[2];
    const int*   e2 = (const int*)d_in[3];
    const int E = in_sizes[2] / 2;

    const float* Wl[5], * bl[5], * Wr[5];
    for (int i = 0; i < 5; ++i) {
        Wl[i] = (const float*)d_in[4 + 3 * i];
        bl[i] = (const float*)d_in[5 + 3 * i];
        Wr[i] = (const float*)d_in[6 + 3 * i];
    }
    const float* mW1 = (const float*)d_in[19];
    const float* mb1 = (const float*)d_in[20];
    const float* mW2 = (const float*)d_in[21];
    const float* mb2 = (const float*)d_in[22];

    size_t off = 0;
    auto carve = [&](size_t bytes) -> char* {
        off = (off + 255) & ~(size_t)255;
        char* p = (char*)d_ws + off;
        off += bytes;
        return p;
    };
    const size_t featB = (size_t)NN * DD * sizeof(float);
    float* H[2], * M[2], * T[2], * cnt[2], * deg[2];
    float* pooled[2], * ssB[2], * caB[2], * csB[2], * trB[2];
    for (int g = 0; g < 2; ++g) {
        H[g] = (float*)carve(featB);
        M[g] = (float*)carve(featB);
        T[g] = (float*)carve(featB);
        cnt[g] = (float*)carve(NN * sizeof(float));
        deg[g] = (float*)carve(NN * sizeof(float));
        pooled[g] = (float*)carve(KK * DD * sizeof(float));
        ssB[g] = (float*)carve(KK * KK * sizeof(float));
        caB[g] = (float*)carve(KK * sizeof(float));
        csB[g] = (float*)carve(KK * sizeof(float));
        trB[g] = (float*)carve(256);
    }
    // bf16 weight copies: [0..9] = (c1..c4: Wl,Wr), [10]=m_W1, [11]=m_W2
    __bf16* Wb[12];
    const float* srcW[12] = {Wl[0], Wr[0], Wl[1], Wr[1], Wl[2], Wr[2],
                             Wl[3], Wr[3], Wl[4], Wr[4], mW1, mW2};
    for (int i = 0; i < 12; ++i) {
        Wb[i] = (__bf16*)carve(DD * DD * sizeof(__bf16));
        convert_bf16_kernel<<<(DD * DD + 255) / 256, 256, 0, stream>>>(srcW[i], Wb[i], DD * DD);
    }

    const int cntBlocks = (E + 255) / 256;
    const int scatBlocks = (E * 32 + 255) / 256;
    const int trBlocks = (E + 7) / 8;

    for (int g = 0; g < 2; ++g) {
        const float* x = g ? x2 : x1;
        const int* src = g ? e2 : e1;
        const int* dst = src + E;
        const int li1 = g ? 1 : 0;   // c1 / c2
        const int li3 = g ? 4 : 3;   // c3 / c4

        (void)hipMemsetAsync(cnt[g], 0, NN * sizeof(float), stream);
        (void)hipMemsetAsync(deg[g], 0, NN * sizeof(float), stream);
        count_kernel<<<cntBlocks, 256, 0, stream>>>(src, dst, cnt[g], deg[g], E);

        // layer 1: H = mish(mean(x) @ Wl + x @ Wr + bl)
        (void)hipMemsetAsync(M[g], 0, featB, stream);
        scatter_kernel<<<scatBlocks, 256, 0, stream>>>(x, src, dst, M[g], E);
        gemm_dual_kernel<true, true, 1><<<NN / 64, 256, 0, stream>>>(
            M[g], cnt[g], x, Wb[2 * li1], Wb[2 * li1 + 1], bl[li1], H[g]);

        // co layer: T = mish(mean(H) @ co_Wl + H @ co_Wr + co_bl)
        (void)hipMemsetAsync(M[g], 0, featB, stream);
        scatter_kernel<<<scatBlocks, 256, 0, stream>>>(H[g], src, dst, M[g], E);
        gemm_dual_kernel<true, true, 1><<<NN / 64, 256, 0, stream>>>(
            M[g], cnt[g], H[g], Wb[4], Wb[5], bl[2], T[g]);

        // residual: T = H + T
        add_kernel<<<NN * DD / 256, 256, 0, stream>>>(H[g], T[g], NN * DD);

        // layer 3: H = mish(mean(T) @ W + T @ Wr + b)
        (void)hipMemsetAsync(M[g], 0, featB, stream);
        scatter_kernel<<<scatBlocks, 256, 0, stream>>>(T[g], src, dst, M[g], E);
        gemm_dual_kernel<true, true, 1><<<NN / 64, 256, 0, stream>>>(
            M[g], cnt[g], T[g], Wb[2 * li3], Wb[2 * li3 + 1], bl[li3], H[g]);

        // DMoN assignments: s = softmax((H@W1+b1)@W2+b2)
        gemm_dual_kernel<false, false, 0><<<NN / 64, 256, 0, stream>>>(
            H[g], nullptr, nullptr, Wb[10], nullptr, mb1, M[g]);
        gemm_dual_kernel<false, false, 0><<<NN / 64, 256, 0, stream>>>(
            M[g], nullptr, nullptr, Wb[11], nullptr, mb2, T[g]);
        softmax_kernel<<<NN, KK, 0, stream>>>(T[g]);  // T = s

        // pooled = s^T @ H ; ss = s^T @ s  (split-K WMMA with f32 atomics)
        (void)hipMemsetAsync(pooled[g], 0, KK * DD * sizeof(float), stream);
        (void)hipMemsetAsync(ssB[g], 0, KK * KK * sizeof(float), stream);
        pool_kernel<<<NN / 64, 256, 0, stream>>>(T[g], H[g], pooled[g]);
        pool_kernel<<<NN / 64, 256, 0, stream>>>(T[g], T[g], ssB[g]);

        (void)hipMemsetAsync(caB[g], 0, KK * sizeof(float), stream);
        (void)hipMemsetAsync(csB[g], 0, KK * sizeof(float), stream);
        (void)hipMemsetAsync(trB[g], 0, sizeof(float), stream);
        colsum_ca_kernel<<<NN / 128, 128, 0, stream>>>(T[g], deg[g], csB[g], caB[g]);
        trace_kernel<<<trBlocks, 256, 0, stream>>>(T[g], src, dst, trB[g], E);
    }

    final_kernel<<<1, 256, 0, stream>>>(
        pooled[0], ssB[0], caB[0], csB[0], trB[0],
        pooled[1], ssB[1], caB[1], csB[1], trB[1],
        (float*)d_out, E);
}
